// MultilHeadAttention_24910810317187
// MI455X (gfx1250) — matrix operand order repaired and checked by reading, not compiled, not run on hardware
//
#include <hip/hip_runtime.h>

typedef __attribute__((ext_vector_type(16))) __bf16 v16bf;
typedef __attribute__((ext_vector_type(8)))  float  v8f;
typedef __attribute__((ext_vector_type(4)))  unsigned int u32x4;
typedef __attribute__((ext_vector_type(8)))  int i32x8;
typedef __attribute__((ext_vector_type(4)))  int i32x4;

#define D_MODEL 1024
#define SEQ     2048
#define BATCH   2
#define HEADS   16
#define HDIM    64
#define MROWS   (BATCH * SEQ) /* 4096 */
#define LDSB    72            /* padded LDS row stride (bf16 elems) = 64 + 8 */

#if defined(__AMDGCN__) && __has_builtin(__builtin_amdgcn_tensor_load_to_lds) && \
    __has_builtin(__builtin_amdgcn_s_wait_tensorcnt)
#define USE_TDM 1
#else
#define USE_TDM 0
#endif

__device__ __forceinline__ v8f vzero8() {
  v8f z;
#pragma unroll
  for (int i = 0; i < 8; ++i) z[i] = 0.0f;
  return z;
}

// A fragment (16x32 bf16, M x K) in contiguous k order (not the ISA table order):
// lanes 0-15: M=lane, K = 0..15; lanes 16-31: M=lane-16, K = 16..31
__device__ __forceinline__ void load_fragA(const __bf16* __restrict__ base, int stride,
                                           int lane, int kc, v16bf& a) {
  const int m  = lane & 15;
  const int kb = ((lane & 16) ? 16 : 0) + kc;
  const __bf16* p = base + m * stride + kb;
#pragma unroll
  for (int i = 0; i < 8; ++i) a[i] = p[i];
#pragma unroll
  for (int i = 0; i < 8; ++i) a[8 + i] = p[8 + i];
}

// B fragment (32x16 bf16, K x N): lanes 0-15: N=lane, K=0..15; lanes 16-31: N=lane-16, K=16..31
// base points at a row-major [16 rows x stride] tile whose row index is N, col index is K.
__device__ __forceinline__ void load_fragB(const __bf16* __restrict__ base, int stride,
                                           int lane, int kc, v16bf& b) {
  const int n  = lane & 15;
  const int kb = ((lane & 16) ? 16 : 0) + kc;
  const __bf16* p = base + n * stride + kb;
#pragma unroll
  for (int i = 0; i < 16; ++i) b[i] = p[i];
}

// C = bf16( x @ W^T + bias ), one weight matrix per blockIdx.z (WQ, WK, WO)
__global__ __launch_bounds__(256) void proj_gemm_kernel(
    const float* __restrict__ x,
    const float* __restrict__ W0, const float* __restrict__ b0,
    const float* __restrict__ W1, const float* __restrict__ b1,
    const float* __restrict__ W2, const float* __restrict__ b2,
    __bf16* __restrict__ O0, __bf16* __restrict__ O1, __bf16* __restrict__ O2)
{
  const float* W; const float* bias; __bf16* Out;
  if (blockIdx.z == 0)      { W = W0; bias = b0; Out = O0; }
  else if (blockIdx.z == 1) { W = W1; bias = b1; Out = O1; }
  else                      { W = W2; bias = b2; Out = O2; }

  __shared__ __align__(16) __bf16 As[128 * LDSB];
  __shared__ __align__(16) __bf16 Bs[128 * LDSB];

  const int tid  = threadIdx.x;
  const int lane = tid & 31;
  const int wid  = tid >> 5;     // 8 waves
  const int wm   = wid >> 2;     // 0..1  (64 M-rows each)
  const int wn   = wid & 3;      // 0..3  (32 N-cols each)
  const int mblk = blockIdx.y * 128;
  const int nblk = blockIdx.x * 128;

  v8f acc[4][2];
#pragma unroll
  for (int i = 0; i < 4; ++i)
#pragma unroll
    for (int j = 0; j < 2; ++j) acc[i][j] = vzero8();

  const int lrow = tid >> 1;       // 0..127
  const int lcol = (tid & 1) * 32; // 0 or 32

  for (int k0 = 0; k0 < D_MODEL; k0 += 64) {
    const float* xs   = x + (size_t)(mblk + lrow) * D_MODEL + k0 + lcol;
    const float* wsrc = W + (size_t)(nblk + lrow) * D_MODEL + k0 + lcol;
    __bf16* ad = As + lrow * LDSB + lcol;
    __bf16* bd = Bs + lrow * LDSB + lcol;
#pragma unroll
    for (int i = 0; i < 32; i += 4) {
      float4 v = *(const float4*)(xs + i);
      ad[i] = (__bf16)v.x; ad[i + 1] = (__bf16)v.y; ad[i + 2] = (__bf16)v.z; ad[i + 3] = (__bf16)v.w;
      float4 w = *(const float4*)(wsrc + i);
      bd[i] = (__bf16)w.x; bd[i + 1] = (__bf16)w.y; bd[i + 2] = (__bf16)w.z; bd[i + 3] = (__bf16)w.w;
    }
    __syncthreads();

#pragma unroll
    for (int kc = 0; kc < 64; kc += 32) {
      v16bf bf[2];
#pragma unroll
      for (int j = 0; j < 2; ++j)
        load_fragB(Bs + (wn * 32 + j * 16) * LDSB, LDSB, lane, kc, bf[j]);
#pragma unroll
      for (int i = 0; i < 4; ++i) {
        v16bf af;
        load_fragA(As + (wm * 64 + i * 16) * LDSB, LDSB, lane, kc, af);
#pragma unroll
        for (int j = 0; j < 2; ++j)
          acc[i][j] = __builtin_amdgcn_wmma_f32_16x16x32_bf16(
              false, af, false, bf[j], (short)0, acc[i][j], false, false);
      }
    }
    __syncthreads();
  }

  const int nlo = lane & 15;
  const int mh  = (lane & 16) ? 8 : 0;
#pragma unroll
  for (int j = 0; j < 2; ++j) {
    const int   ng  = nblk + wn * 32 + j * 16 + nlo;
    const float bsv = bias[ng];
#pragma unroll
    for (int i = 0; i < 4; ++i)
#pragma unroll
      for (int v = 0; v < 8; ++v) {
        const int mg = mblk + wm * 64 + i * 16 + v + mh;
        Out[(size_t)mg * D_MODEL + ng] = (__bf16)(acc[i][j][v] + bsv);
      }
  }
}

// Flash attention over contiguous (2048, 64) per-(b,h) views of Q/K/V.
__global__ __launch_bounds__(256) void flash_attn_kernel(
    const __bf16* __restrict__ Q, const __bf16* __restrict__ K,
    const __bf16* __restrict__ V, float* __restrict__ out)
{
  __shared__ __align__(16) __bf16 Ks[64 * LDSB];       // [kv][d]
  __shared__ __align__(16) __bf16 Vt[64 * LDSB];       // [d][kv] (transposed)
  __shared__ __align__(16) __bf16 Ps[8 * 16 * LDSB];   // per-wave P tile [16][64]

  const int tid  = threadIdx.x;
  const int lane = tid & 31;
  const int wid  = tid >> 5;
  const int qt = blockIdx.x, h = blockIdx.y, b = blockIdx.z;

  const size_t hoff = ((size_t)b * SEQ + (size_t)h * 128) * D_MODEL;
  const __bf16* Qh = Q + hoff;  // (2048, 64) row-major
  const __bf16* Kh = K + hoff;
  const __bf16* Vh = V + hoff;

  const int qrow0 = qt * 128 + wid * 16;

  v16bf q0, q1;
  load_fragA(Qh + (size_t)qrow0 * HDIM, HDIM, lane, 0,  q0);
  load_fragA(Qh + (size_t)qrow0 * HDIM, HDIM, lane, 32, q1);

  float mrow[8], lsum[8];
  v8f o[4];
#pragma unroll
  for (int v = 0; v < 8; ++v) { mrow[v] = -1e30f; lsum[v] = 0.0f; }
#pragma unroll
  for (int nt = 0; nt < 4; ++nt) o[nt] = vzero8();

  const float scale = 1.0f / 64.0f;  // reference divides by head_dim, NOT sqrt
  const int lr = tid >> 2;           // 0..63
  const int lc = (tid & 3) * 16;     // 0,16,32,48

  __bf16* Pw = Ps + wid * 16 * LDSB;
  const int nlo = lane & 15;
  const int mh  = (lane & 16) ? 8 : 0;

#if USE_TDM
  // TDM descriptor invariants for the K tile:
  //  - 64x64 bf16 tile, contiguous in global (tile start = Kh + kv0*64)
  //  - LDS padding: +16B every 128B  ->  row stride 72 bf16 (pad_interval enc 4 = 32 DW,
  //    pad_amount enc 3 = 4 DW)
  const unsigned int ldsK = (unsigned int)(unsigned long long)(const void*)Ks;
  i32x8 g1;
  g1[0] = (1 << 16)          // data_size = 2 bytes
        | (1 << 20)          // pad_enable
        | (4 << 22)          // pad_interval: 32 DWORDs = 128B
        | (3 << 25);         // pad_amount:   4 DWORDs = 16B
  g1[1] = (HDIM << 16);      // atomic_barrier_addr=0 ; tensor_dim0[15:0] = 64
  g1[2] = (SEQ << 16);       // tensor_dim0[31:16]=0 ; tensor_dim1[15:0] = 2048
  g1[3] = (HDIM << 16);      // tensor_dim1[31:16]=0 ; tile_dim0 = 64
  g1[4] = 64;                // tile_dim1 = 64, tile_dim2 = 0
  g1[5] = HDIM;              // tensor_dim0_stride[31:0] = 64 elems
  g1[6] = 0;
  g1[7] = 0;
  i32x4 gnull4;
  gnull4[0] = 0; gnull4[1] = 0; gnull4[2] = 0; gnull4[3] = 0;
  i32x8 gnull8;
#pragma unroll
  for (int i = 0; i < 8; ++i) gnull8[i] = 0;
#endif

  for (int kv0 = 0; kv0 < SEQ; kv0 += 64) {
#if USE_TDM
    if (wid == 0) { // one wave drives the Tensor Data Mover (EXEC is ignored by TDM)
      const unsigned long long ga =
          (unsigned long long)(const void*)(Kh + (size_t)kv0 * HDIM);
      u32x4 g0;
      g0[0] = 1u;                                   // count=1, user descriptor
      g0[1] = ldsK;                                 // lds_addr
      g0[2] = (unsigned int)ga;                     // global_addr[31:0]
      g0[3] = (unsigned int)((ga >> 32) & 0x1FFFFFFu) | (2u << 30); // addr[56:32] | type=2
      __builtin_amdgcn_tensor_load_to_lds(g0, g1, gnull4, gnull4, gnull8, 0);
    }
#else
    { // K tile, row-major
      const __bf16* src = Kh + (size_t)(kv0 + lr) * HDIM + lc;
      __bf16* dst = Ks + lr * LDSB + lc;
      ((uint4*)dst)[0] = ((const uint4*)src)[0];
      ((uint4*)dst)[1] = ((const uint4*)src)[1];
    }
#endif
    { // V tile, transposed into Vt[d][kv]
      const __bf16* src = Vh + (size_t)(kv0 + lr) * HDIM + lc;
#pragma unroll
      for (int i = 0; i < 16; ++i) Vt[(lc + i) * LDSB + lr] = src[i];
    }
    if (kv0 + 64 < SEQ) {
      __builtin_prefetch(Kh + (size_t)(kv0 + 64 + lr) * HDIM + lc, 0, 1);
      __builtin_prefetch(Vh + (size_t)(kv0 + 64 + lr) * HDIM + lc, 0, 1);
    }
#if USE_TDM
    if (wid == 0) __builtin_amdgcn_s_wait_tensorcnt(0);
#endif
    __syncthreads();

    // S = (Q K^T) / 64 : 4 N-tiles of 16 kv cols, K-dim = 64 (two wmma)
    v8f s[4];
#pragma unroll
    for (int jt = 0; jt < 4; ++jt) {
      v8f a = vzero8();
      v16bf kb0, kb1;
      load_fragB(Ks + jt * 16 * LDSB, LDSB, lane, 0,  kb0);
      load_fragB(Ks + jt * 16 * LDSB, LDSB, lane, 32, kb1);
      a = __builtin_amdgcn_wmma_f32_16x16x32_bf16(false, q0, false, kb0, (short)0, a, false, false);
      a = __builtin_amdgcn_wmma_f32_16x16x32_bf16(false, q1, false, kb1, (short)0, a, false, false);
#pragma unroll
      for (int v = 0; v < 8; ++v) s[jt][v] = a[v] * scale;
    }

    // online softmax; C-layout row v (+8 for upper half) spans 16 lanes -> xor reduce
    float tmax[8];
#pragma unroll
    for (int v = 0; v < 8; ++v) {
      float m = s[0][v];
#pragma unroll
      for (int jt = 1; jt < 4; ++jt) m = fmaxf(m, s[jt][v]);
      tmax[v] = m;
    }
#pragma unroll
    for (int msk = 1; msk <= 8; msk <<= 1)
#pragma unroll
      for (int v = 0; v < 8; ++v) tmax[v] = fmaxf(tmax[v], __shfl_xor(tmax[v], msk, 32));

    float f[8];
#pragma unroll
    for (int v = 0; v < 8; ++v) {
      const float mn = fmaxf(mrow[v], tmax[v]);
      f[v] = __expf(mrow[v] - mn);
      mrow[v] = mn;
    }

    float tsum[8];
#pragma unroll
    for (int v = 0; v < 8; ++v) tsum[v] = 0.0f;
#pragma unroll
    for (int jt = 0; jt < 4; ++jt)
#pragma unroll
      for (int v = 0; v < 8; ++v) {
        const float p = __expf(s[jt][v] - mrow[v]);
        s[jt][v] = p;
        tsum[v] += p;
      }
#pragma unroll
    for (int msk = 1; msk <= 8; msk <<= 1)
#pragma unroll
      for (int v = 0; v < 8; ++v) tsum[v] += __shfl_xor(tsum[v], msk, 32);
#pragma unroll
    for (int v = 0; v < 8; ++v) lsum[v] = lsum[v] * f[v] + tsum[v];
#pragma unroll
    for (int nt = 0; nt < 4; ++nt)
#pragma unroll
      for (int v = 0; v < 8; ++v) o[nt][v] *= f[v];

    // P (C layout) -> LDS -> A layout for P @ V
#pragma unroll
    for (int jt = 0; jt < 4; ++jt)
#pragma unroll
      for (int v = 0; v < 8; ++v)
        Pw[(v + mh) * LDSB + jt * 16 + nlo] = (__bf16)s[jt][v];

    asm volatile("s_wait_dscnt 0" ::: "memory");

#pragma unroll
    for (int kc4 = 0; kc4 < 64; kc4 += 32) {
      v16bf pa;
      load_fragA(Pw, LDSB, lane, kc4, pa);
#pragma unroll
      for (int nt = 0; nt < 4; ++nt) {
        v16bf vb;
        load_fragB(Vt + nt * 16 * LDSB, LDSB, lane, kc4, vb);
        o[nt] = __builtin_amdgcn_wmma_f32_16x16x32_bf16(false, pa, false, vb, (short)0, o[nt], false, false);
      }
    }
    __syncthreads();
  }

  // out[b, sq, h*64 + d] = O / l   (undoes the transpose(1,2)+reshape)
#pragma unroll
  for (int nt = 0; nt < 4; ++nt)
#pragma unroll
    for (int v = 0; v < 8; ++v) {
      const int sq  = qrow0 + v + mh;
      const int col = h * HDIM + nt * 16 + nlo;
      out[(size_t)b * SEQ * D_MODEL + (size_t)sq * D_MODEL + col] = o[nt][v] / lsum[v];
    }
}

extern "C" void kernel_launch(void* const* d_in, const int* in_sizes, int n_in,
                              void* d_out, int out_size, void* d_ws, size_t ws_size,
                              hipStream_t stream) {
  (void)in_sizes; (void)n_in; (void)out_size; (void)ws_size;
  const float* x  = (const float*)d_in[0];
  const float* WQ = (const float*)d_in[1];
  const float* bQ = (const float*)d_in[2];
  const float* WK = (const float*)d_in[3];
  const float* bK = (const float*)d_in[4];
  const float* WO = (const float*)d_in[5];
  const float* bO = (const float*)d_in[6];
  float* out = (float*)d_out;

  const size_t projBytes = (size_t)MROWS * D_MODEL * sizeof(__bf16); // 8 MB each
  __bf16* Qb = (__bf16*)d_ws;
  __bf16* Kb = (__bf16*)((char*)d_ws + projBytes);
  __bf16* Vb = (__bf16*)((char*)d_ws + 2 * projBytes);

  dim3 gp(D_MODEL / 128, MROWS / 128, 3);
  proj_gemm_kernel<<<gp, dim3(256), 0, stream>>>(x, WQ, bQ, WK, bK, WO, bO, Qb, Kb, Vb);

  dim3 ga(SEQ / 128, HEADS, BATCH);
  flash_attn_kernel<<<ga, dim3(256), 0, stream>>>(Qb, Kb, Vb, out);
}
